// MS_SSIM_L1_LOSS_2164663517262
// MI455X (gfx1250) — matrix operand order repaired and checked by reading, not compiled, not run on hardware
//
#include <hip/hip_runtime.h>
#include <math.h>

typedef __attribute__((ext_vector_type(16))) _Float16       v16h;
typedef __attribute__((ext_vector_type(8)))  _Float16       v8h;
typedef __attribute__((ext_vector_type(8)))  unsigned short v8u;
typedef __attribute__((ext_vector_type(8)))  float          v8f;

#define MS_H 512
#define MS_W 512
#define MS_NB 8
#define MS_C1v 0.0001f
#define MS_C2v 0.0009f
#define MS_ALPHA 0.025f
#define MS_COMP 200.0f

__device__ __forceinline__ v8f wmma16(v16h a, v16h b, v8f c) {
    // v_wmma_f32_16x16x32_f16: D = A(16x32) * B(32x16) + C
    return __builtin_amdgcn_wmma_f32_16x16x32_f16(false, a, false, b, (short)0, c,
                                                  false, false);
}

// Toeplitz fragments for taps tp[0..32]:
//  B  (32x16, B layout):  B[k][n]  = g[k-n]   (horizontal pass operand)
//  AT (16x32, A layout):  AT[m][k] = g[k-m]   (vertical pass operand)
__device__ __forceinline__ void build_toeplitz(const float* __restrict__ tp, int lane,
                                               v16h& B0, v16h& B1, v16h& AT0, v16h& AT1) {
    const int n    = lane & 15;
    const int kh16 = (lane >> 4) * 16;
    const int kh8  = (lane >> 4) * 8;
#pragma unroll
    for (int e = 0; e < 16; ++e) {
        const int kb = kh16 + e;
        const int ka = kh8 + (e & 7) + ((e & 8) ? 16 : 0);
        const int d0 = ka - n;
        const int d1 = ka + 32 - n;
        B0[e] = ((unsigned)d0 <= 32u) ? (_Float16)tp[d0] : (_Float16)0.f;
        B1[e] = ((unsigned)d1 <= 32u) ? (_Float16)tp[d1] : (_Float16)0.f;
        const int e0 = kb - n;
        const int e1 = kb + 32 - n;
        AT0[e] = ((unsigned)e0 <= 32u) ? (_Float16)tp[e0] : (_Float16)0.f;
        AT1[e] = ((unsigned)e1 <= 32u) ? (_Float16)tp[e1] : (_Float16)0.f;
    }
}

// Packed-f16 field evaluation (v_pk_mul_f16 / v_pk_add_f16, sign-mask abs).
template <int F>
__device__ __forceinline__ v8h fv8(v8h xv, v8h yv) {
    if constexpr (F == 0) return xv;
    else if constexpr (F == 1) return yv;
    else if constexpr (F == 2) return xv * xv;
    else if constexpr (F == 3) return yv * yv;
    else if constexpr (F == 4) return xv * yv;
    else {
        v8h  d = xv - yv;
        v8u  u = __builtin_bit_cast(v8u, d);
        u = u & (unsigned short)0x7FFF;
        return __builtin_bit_cast(v8h, u);
    }
}

// Horizontal pass of one field: all 8 waves, chunk (rc,cc) of 64 rows x 32 cols.
template <int F>
__device__ __forceinline__ void hpass(const _Float16* __restrict__ xt,
                                      const _Float16* __restrict__ yt,
                                      _Float16* __restrict__ hb,
                                      const v16h& B0h, const v16h& B1h,
                                      int wave, int lane) {
    const int rc   = wave >> 1, cc = wave & 1;
    const int row  = rc * 16 + (lane & 15);
    const int colb = cc * 16 + (lane >> 4) * 8;
    const _Float16* xb = xt + row * 80 + colb;
    const _Float16* yb = yt + row * 80 + colb;

    constexpr bool needx = (F != 1 && F != 3);
    constexpr bool needy = (F != 0 && F != 2);
    v8h x0{}, x1{}, x2{}, x3{}, y0{}, y1{}, y2{}, y3{};
    if constexpr (needx) {
        x0 = *(const v8h*)(xb);      x1 = *(const v8h*)(xb + 16);
        x2 = *(const v8h*)(xb + 32); x3 = *(const v8h*)(xb + 48);
    }
    if constexpr (needy) {
        y0 = *(const v8h*)(yb);      y1 = *(const v8h*)(yb + 16);
        y2 = *(const v8h*)(yb + 32); y3 = *(const v8h*)(yb + 48);
    }
    const v8h a0lo = fv8<F>(x0, y0), a0hi = fv8<F>(x1, y1);
    const v8h a1lo = fv8<F>(x2, y2), a1hi = fv8<F>(x3, y3);
    const v16h A0 = __builtin_shufflevector(a0lo, a0hi, 0, 1, 2, 3, 4, 5, 6, 7,
                                            8, 9, 10, 11, 12, 13, 14, 15);
    const v16h A1 = __builtin_shufflevector(a1lo, a1hi, 0, 1, 2, 3, 4, 5, 6, 7,
                                            8, 9, 10, 11, 12, 13, 14, 15);
    v8f d = {0.f, 0.f, 0.f, 0.f, 0.f, 0.f, 0.f, 0.f};
    d = wmma16(A0, B0h, d);
    d = wmma16(A1, B1h, d);
    const v8h hs = __builtin_convertvector(d, v8h);
    // column-major store: lane owns one column, 8 consecutive rows
    *(v8h*)(hb + (cc * 16 + (lane & 15)) * 72 + rc * 16 + (lane >> 4) * 8) = hs;
}

// Vertical pass of one field: one wave-quad, 2x2 chunks of the 32x32 output.
__device__ __forceinline__ void vpass(const _Float16* __restrict__ hb,
                                      float* __restrict__ oslot,
                                      const v16h& AT0, const v16h& AT1,
                                      int wq, int lane) {
    const int rc2  = wq >> 1, cc2 = wq & 1;
    const int colh = cc2 * 16 + (lane & 15);
    const int kb   = (lane >> 4) * 16;
    const _Float16* p = hb + colh * 72 + rc2 * 16 + kb;
    const v8h t0 = *(const v8h*)(p);
    const v8h t1 = *(const v8h*)(p + 8);
    const v8h t2 = *(const v8h*)(p + 32);
    const v8h t3 = *(const v8h*)(p + 40);
    const v16h Bv0 = __builtin_shufflevector(t0, t1, 0, 1, 2, 3, 4, 5, 6, 7,
                                             8, 9, 10, 11, 12, 13, 14, 15);
    const v16h Bv1 = __builtin_shufflevector(t2, t3, 0, 1, 2, 3, 4, 5, 6, 7,
                                             8, 9, 10, 11, 12, 13, 14, 15);
    v8f d2 = {0.f, 0.f, 0.f, 0.f, 0.f, 0.f, 0.f, 0.f};
    d2 = wmma16(AT0, Bv0, d2);
    d2 = wmma16(AT1, Bv1, d2);
    float* ob = oslot + (rc2 * 16 + (lane >> 4) * 8) * 33 + cc2 * 16 + (lane & 15);
#pragma unroll
    for (int i = 0; i < 8; ++i) ob[i * 33] = d2[i];
}

__global__ void __launch_bounds__(256)
msssim_l1_kernel(const float* __restrict__ x, const float* __restrict__ y,
                 const float* __restrict__ gm, float* __restrict__ out) {
    __shared__ __align__(16) _Float16 xt[64 * 80];     // channel tile f16 + zero apron
    __shared__ __align__(16) _Float16 yt[64 * 80];
    __shared__ __align__(16) _Float16 Hb[2][32 * 72];  // double-buffered H-blur (col-major)
    __shared__ float o32[6 * 32 * 33];                 // vertical results (f32)
    __shared__ float g1d[5 * 33];                      // separable taps per sigma
    __shared__ float red[256];

    const int tid  = threadIdx.x;
    const int lane = tid & 31;
    const int wave = tid >> 5;
    const int bx   = blockIdx.x;
    const int b    = bx >> 8;
    const int tile = bx & 255;
    const int r0   = (tile >> 4) * 32;
    const int c0   = (tile & 15) * 32;

    // Recover 1D taps: G2d = outer(g,g) => g[j] = G[16][j] / sqrt(G[16][16])
    if (tid < 5 * 33) {
        const int s = tid / 33, j = tid % 33;
        const float* grow = gm + ((size_t)(3 * s) * 33 + 16) * 33;
        g1d[s * 33 + j] = grow[j] * __frsqrt_rn(grow[16]);
    }

    float PIcs[4] = {1.f, 1.f, 1.f, 1.f};
    float lM[4]   = {0.f, 0.f, 0.f, 0.f};
    float gl1[4]  = {0.f, 0.f, 0.f, 0.f};

    // (channel, sigma) combos from the grouped-conv mapping.
    // flags: 1 = SSIM (5 fields), 2 = L1 conv (|x-y|), 4 = luminance term
    const signed char combo_c[9]  = {0, 0, 0, 1, 1, 1, 1, 2, 2};
    const signed char combo_s[9]  = {0, 1, 4, 1, 2, 3, 4, 3, 4};
    const signed char combo_m[9]  = {3, 2, 0, 1, 3, 1, 0, 2, 3};
    const signed char combo_fl[9] = {1, 1, 2, 1, 1, 1, 2, 1, 7};

    int prev_c = -1;
#pragma unroll 1
    for (int ci = 0; ci < 9; ++ci) {
        const int c = combo_c[ci], s = combo_s[ci];
        const int mlt = combo_m[ci], fl = combo_fl[ci];

        if (c != prev_c) {
            __syncthreads();
            const size_t chb = ((size_t)(b * 3 + c)) * MS_H * MS_W;
#pragma unroll 1
            for (int idx = tid; idx < 64 * 64; idx += 256) {
                const int rr = idx >> 6, ccl = idx & 63;
                const int gr = r0 - 16 + rr, gc = c0 - 16 + ccl;
                float xv = 0.f, yv = 0.f;
                if ((unsigned)gr < MS_H && (unsigned)gc < MS_W) {
                    const size_t o = chb + (size_t)gr * MS_W + gc;
                    xv = x[o];
                    yv = y[o];
                    if (c < 2) {  // warm L2/WGP$ for the next channel (global_prefetch_b8)
                        __builtin_prefetch(x + o + (size_t)MS_H * MS_W, 0, 1);
                        __builtin_prefetch(y + o + (size_t)MS_H * MS_W, 0, 1);
                    }
                }
                xt[rr * 80 + ccl] = (_Float16)xv;
                yt[rr * 80 + ccl] = (_Float16)yv;
            }
#pragma unroll 1
            for (int idx = tid; idx < 64 * 16; idx += 256) {  // zero apron
                const int rr = idx >> 4, ccl = 64 + (idx & 15);
                xt[rr * 80 + ccl] = (_Float16)0.f;
                yt[rr * 80 + ccl] = (_Float16)0.f;
            }
            prev_c = c;
            __syncthreads();
        }

        v16h B0h, B1h, AT0, AT1;
        build_toeplitz(&g1d[s * 33], lane, B0h, B1h, AT0, AT1);

        int fields[6];
        int nf = 0;
        if (fl & 1) {
            fields[0] = 0; fields[1] = 1; fields[2] = 2; fields[3] = 3; fields[4] = 4;
            nf = 5;
        }
        if (fl & 2) fields[nf++] = 5;

        // Software-pipelined phases: H(f) on all 8 waves overlapped with V(f-1)
        // on the alternating wave-quad from the other Hbuf.
#pragma unroll 1
        for (int ph = 0; ph <= nf; ++ph) {
            __syncthreads();
            if (ph < nf) {
                _Float16* hb = Hb[ph & 1];
                switch (fields[ph]) {
                    case 0: hpass<0>(xt, yt, hb, B0h, B1h, wave, lane); break;
                    case 1: hpass<1>(xt, yt, hb, B0h, B1h, wave, lane); break;
                    case 2: hpass<2>(xt, yt, hb, B0h, B1h, wave, lane); break;
                    case 3: hpass<3>(xt, yt, hb, B0h, B1h, wave, lane); break;
                    case 4: hpass<4>(xt, yt, hb, B0h, B1h, wave, lane); break;
                    default: hpass<5>(xt, yt, hb, B0h, B1h, wave, lane); break;
                }
            }
            if (ph > 0) {
                const int par = (ph - 1) & 1;
                if ((wave >> 2) == par) {  // alternate which quad does vertical work
                    vpass(Hb[par], &o32[fields[ph - 1] * 1056], AT0, AT1, wave & 3, lane);
                }
            }
        }
        __syncthreads();  // o32 complete

#pragma unroll
        for (int px = 0; px < 4; ++px) {
            const int p  = tid + px * 256;
            const int pb = (p >> 5) * 33 + (p & 31);
            if (fl & 1) {
                const float mux = o32[0 * 1056 + pb];
                const float muy = o32[1 * 1056 + pb];
                const float ex2 = o32[2 * 1056 + pb];
                const float ey2 = o32[3 * 1056 + pb];
                const float exy = o32[4 * 1056 + pb];
                const float mux2 = mux * mux, muy2 = muy * muy, muxy = mux * muy;
                const float sx2 = ex2 - mux2, sy2 = ey2 - muy2, sxy = exy - muxy;
                const float l  = (2.f * muxy + MS_C1v) / (mux2 + muy2 + MS_C1v);
                const float cs = (2.f * sxy + MS_C2v) / (sx2 + sy2 + MS_C2v);
                float csm = cs;
                if (mlt >= 2) csm *= cs;
                if (mlt >= 3) csm *= cs;
                PIcs[px] *= csm;
                if (fl & 4) lM[px] = l * l * l;
            }
            if (fl & 2) gl1[px] += o32[5 * 1056 + pb] * (1.f / 3.f);
        }
    }

    float local = 0.f;
#pragma unroll
    for (int px = 0; px < 4; ++px)
        local += MS_ALPHA * (1.f - lM[px] * PIcs[px]) + (1.f - MS_ALPHA) * gl1[px];
    local *= MS_COMP / (float)(MS_NB * MS_H * MS_W);

    red[tid] = local;
    __syncthreads();
#pragma unroll
    for (int st = 128; st > 0; st >>= 1) {
        if (tid < st) red[tid] += red[tid + st];
        __syncthreads();
    }
    if (tid == 0) atomicAdd(out, red[0]);
}

__global__ void msssim_zero_out(float* __restrict__ out) {
    if (threadIdx.x == 0 && blockIdx.x == 0) out[0] = 0.f;
}

extern "C" void kernel_launch(void* const* d_in, const int* in_sizes, int n_in,
                              void* d_out, int out_size, void* d_ws, size_t ws_size,
                              hipStream_t stream) {
    (void)in_sizes; (void)n_in; (void)out_size; (void)d_ws; (void)ws_size;
    const float* x  = (const float*)d_in[0];
    const float* y  = (const float*)d_in[1];
    const float* gm = (const float*)d_in[2];
    float* out = (float*)d_out;

    msssim_zero_out<<<1, 32, 0, stream>>>(out);
    msssim_l1_kernel<<<MS_NB * 256, 256, 0, stream>>>(x, y, gm, out);
}
